// LmHeadTender_8117488190107
// MI455X (gfx1250) — 8-bit matrix operand order repaired and checked by reading, compiled, not run on hardware
//
#include <hip/hip_runtime.h>


typedef int   v16i __attribute__((ext_vector_type(16)));
typedef float v8f  __attribute__((ext_vector_type(8)));
typedef __attribute__((address_space(3))) unsigned char lds_uchar;

#if __has_builtin(__builtin_amdgcn_wmma_f32_16x16x128_bf8_bf8)
#define WMMA128(a, b, c) \
  __builtin_amdgcn_wmma_f32_16x16x128_bf8_bf8((a), (b), (short)0, (c), false, false)
#else
#define WMMA128(a, b, c) \
  __builtin_amdgcn_wmma_f32_16x16x128_fp8_fp8((a), (b), (short)0, (c), false, false)
#endif

#define HID   4096
#define VOCAB 32000
#define TOK   4096
#define NCHUNK 16
#define BM 128
#define BN 128
#define BK 128
#define NBUF 3

__device__ __forceinline__ void async_load16(lds_uchar* dst,
                                             const unsigned char* src) {
  asm volatile("global_load_async_to_lds_b128 %0, %1, off"
               :: "v"(dst), "v"(src)
               : "memory");
}

__device__ __forceinline__ void wait_async0() {
#if __has_builtin(__builtin_amdgcn_s_wait_asynccnt)
  __builtin_amdgcn_s_wait_asynccnt(0);
#else
  asm volatile("s_wait_asynccnt 0x0" ::: "memory");
#endif
}

__device__ __forceinline__ void wait_async4() {
#if __has_builtin(__builtin_amdgcn_s_wait_asynccnt)
  __builtin_amdgcn_s_wait_asynccnt(4);
#else
  asm volatile("s_wait_asynccnt 0x4" ::: "memory");
#endif
}

__device__ __forceinline__ unsigned char q_to_bf8(int q, int eshift) {
  int m = q < 0 ? -q : q;
  if (m == 0) return 0;
  int e    = 31 - __clz(m);
  int mant = ((m << 2) >> e) & 3;
  int byte = ((e + 15 + eshift) << 2) | mant;
  if (q < 0) byte |= 0x80;
  return (unsigned char)byte;
}

__device__ __forceinline__ float blockMax256(float v) {
  __shared__ float red[8];
  for (int o = 16; o; o >>= 1) v = fmaxf(v, __shfl_down(v, o, 32));
  if ((threadIdx.x & 31) == 0) red[threadIdx.x >> 5] = v;
  __syncthreads();
  if (threadIdx.x == 0) {
    float r = red[0];
    for (int i = 1; i < 8; ++i) r = fmaxf(r, red[i]);
    red[0] = r;
  }
  __syncthreads();
  return red[0];
}

__global__ __launch_bounds__(256) void wquant_kernel(
    const float* __restrict__ W, unsigned char* __restrict__ Wq,
    float* __restrict__ sw) {
  const size_t v = blockIdx.x;
  const float* row = W + v * HID;
  float m = 0.f;
  for (int i0 = threadIdx.x * 4; i0 < HID; i0 += 1024) {
    float4 x = *(const float4*)(row + i0);
    m = fmaxf(m, fmaxf(fmaxf(fabsf(x.x), fabsf(x.y)),
                       fmaxf(fabsf(x.z), fabsf(x.w))));
  }
  m = blockMax256(m);
  const float s = fmaxf(m * (1.f / 7.f), 1e-9f);
  if (threadIdx.x == 0) sw[v] = s;
  const float inv = 1.f / s;
  for (int i0 = threadIdx.x * 4; i0 < HID; i0 += 1024) {
    float4 x = *(const float4*)(row + i0);
    int q0 = min(7, max(-8, (int)rintf(x.x * inv)));
    int q1 = min(7, max(-8, (int)rintf(x.y * inv)));
    int q2 = min(7, max(-8, (int)rintf(x.z * inv)));
    int q3 = min(7, max(-8, (int)rintf(x.w * inv)));
    uchar4 b;
    b.x = q_to_bf8(q0, 0); b.y = q_to_bf8(q1, 0);
    b.z = q_to_bf8(q2, 0); b.w = q_to_bf8(q3, 0);
    *(uchar4*)(Wq + v * HID + i0) = b;
  }
}

__global__ __launch_bounds__(256) void cmax_kernel(
    const float* __restrict__ X, float* __restrict__ cmax) {
  const int c = blockIdx.x;
  const int col0 = blockIdx.y * 1024 + threadIdx.x * 4;
  float4 m = make_float4(0.f, 0.f, 0.f, 0.f);
  for (int r = 0; r < 256; ++r) {
    float4 x = *(const float4*)(X + ((size_t)(c * 256 + r)) * HID + col0);
    m.x = fmaxf(m.x, fabsf(x.x)); m.y = fmaxf(m.y, fabsf(x.y));
    m.z = fmaxf(m.z, fabsf(x.z)); m.w = fmaxf(m.w, fabsf(x.w));
  }
  *(float4*)(cmax + (size_t)c * HID + col0) = m;
}

__global__ __launch_bounds__(256) void tmax_kernel(
    const float* __restrict__ cmax, float* __restrict__ tmax) {
  const int c = blockIdx.x;
  float m = 0.f;
  for (int i = threadIdx.x; i < HID; i += 256)
    m = fmaxf(m, cmax[(size_t)c * HID + i]);
  m = blockMax256(m);
  if (threadIdx.x == 0) tmax[c] = m;
}

__global__ __launch_bounds__(256) void xquant_kernel(
    const float* __restrict__ X, const float* __restrict__ cmax,
    const float* __restrict__ tmax, unsigned char* __restrict__ Xq) {
  const int c = blockIdx.x;
  const int col0 = blockIdx.y * 1024 + threadIdx.x * 4;
  const float tm = tmax[c];
  float4 cm = *(const float4*)(cmax + (size_t)c * HID + col0);
  float cmv[4] = {cm.x, cm.y, cm.z, cm.w};
  int es[4];
  float inv[4];
  #pragma unroll
  for (int j = 0; j < 4; ++j) {
    int bucket = 0;
    #pragma unroll
    for (int i = 0; i < 13; ++i)
      bucket += (cmv[j] > ldexpf(tm, i - 13)) ? 1 : 0;
    es[j] = bucket - 13;
    float s = fmaxf(ldexpf(tm, es[j]) * (1.f / 7.f), 1e-9f);
    inv[j] = 1.f / s;
  }
  for (int r = 0; r < 256; ++r) {
    const size_t rowoff = ((size_t)(c * 256 + r)) * HID + col0;
    float4 x = *(const float4*)(X + rowoff);
    int q0 = min(7, max(-8, (int)rintf(x.x * inv[0])));
    int q1 = min(7, max(-8, (int)rintf(x.y * inv[1])));
    int q2 = min(7, max(-8, (int)rintf(x.z * inv[2])));
    int q3 = min(7, max(-8, (int)rintf(x.w * inv[3])));
    uchar4 b;
    b.x = q_to_bf8(q0, es[0]); b.y = q_to_bf8(q1, es[1]);
    b.z = q_to_bf8(q2, es[2]); b.w = q_to_bf8(q3, es[3]);
    *(uchar4*)(Xq + rowoff) = b;
  }
}

__global__ __launch_bounds__(512) void gemm_bf8_kernel(
    const unsigned char* __restrict__ Aq, const unsigned char* __restrict__ Bq,
    const float* __restrict__ sw, const float* __restrict__ tmax,
    float* __restrict__ out) {
  __shared__ unsigned char lA[NBUF][BM * BK];
  __shared__ unsigned char lB[NBUF][BN * BK];

  const int tid   = threadIdx.x;
  const int mTile = blockIdx.x;
  const int nTile = blockIdx.y;
  const int lane  = tid & 31;
  const int wid   = tid >> 5;
  const int wm    = wid >> 2;
  const int wn    = wid & 3;
  const int lsub  = lane & 15;
  const int hi    = lane >> 4;

  const int r0  = tid >> 3;
  const int c0  = (tid & 7) * 16;
  const int idx = r0 * BK + c0;
  const unsigned char* pA0 =
      Aq + (size_t)(mTile * BM + r0) * HID + c0;
  const unsigned char* pB0 =
      Bq + (size_t)(nTile * BN + r0) * HID + c0;

  auto issue = [&](int kk, int buf) {
    const size_t kOff = (size_t)kk * BK;
    async_load16((lds_uchar*)&lA[buf][idx], pA0 + kOff);
    async_load16((lds_uchar*)&lA[buf][idx + 64 * BK], pA0 + (size_t)64 * HID + kOff);
    async_load16((lds_uchar*)&lB[buf][idx], pB0 + kOff);
    async_load16((lds_uchar*)&lB[buf][idx + 64 * BK], pB0 + (size_t)64 * HID + kOff);
  };

  v8f acc[2][2];
  #pragma unroll
  for (int mf = 0; mf < 2; ++mf)
    #pragma unroll
    for (int nf = 0; nf < 2; ++nf)
      #pragma unroll
      for (int r = 0; r < 8; ++r) acc[mf][nf][r] = 0.f;

  issue(0, 0);
  issue(1, 1);
  wait_async4();
  __syncthreads();

  const int KT = HID / BK;
  int buf = 0;
  for (int k = 0; k < KT; ++k) {
    const int bufN2 = (buf + 2 >= NBUF) ? buf + 2 - NBUF : buf + 2;
    if (k + 2 < KT) issue(k + 2, bufN2);

    v16i bfrag[2];
    #pragma unroll
    for (int nf = 0; nf < 2; ++nf) {
      const unsigned char* bp =
          &lB[buf][(wn * 32 + nf * 16 + lsub) * BK + hi * 16];
      #pragma unroll
      for (int g = 0; g < 4; ++g) {
        uint4 t = *(const uint4*)(bp + g * 32);
        bfrag[nf][4 * g + 0] = (int)t.x;
        bfrag[nf][4 * g + 1] = (int)t.y;
        bfrag[nf][4 * g + 2] = (int)t.z;
        bfrag[nf][4 * g + 3] = (int)t.w;
      }
    }

    #pragma unroll
    for (int mf = 0; mf < 2; ++mf) {
      v16i afrag;
      const unsigned char* ap =
          &lA[buf][(wm * 32 + mf * 16 + lsub) * BK + hi * 16];
      #pragma unroll
      for (int j = 0; j < 8; ++j) {
        uint2 t = *(const uint2*)(ap + (j >> 1) * 32 + (j & 1) * 8);
        afrag[2 * j + 0] = (int)t.x;
        afrag[2 * j + 1] = (int)t.y;
      }
      acc[mf][0] = WMMA128(afrag, bfrag[0], acc[mf][0]);
      acc[mf][1] = WMMA128(afrag, bfrag[1], acc[mf][1]);
    }

    if (k + 2 < KT)      wait_async4();
    else if (k + 1 < KT) wait_async0();
    __syncthreads();
    buf = (buf + 1 >= NBUF) ? 0 : buf + 1;
  }

  const int chunk = (mTile * BM) >> 8;
  const float rs = tmax[chunk] * (1.f / 7.f);
  #pragma unroll
  for (int nf = 0; nf < 2; ++nf) {
    const int col = nTile * BN + wn * 32 + nf * 16 + lsub;
    const float cs = sw[col] * rs;
    #pragma unroll
    for (int mf = 0; mf < 2; ++mf) {
      const int rowb = mTile * BM + wm * 32 + mf * 16 + hi * 8;
      #pragma unroll
      for (int r = 0; r < 8; ++r)
        out[(size_t)(rowb + r) * VOCAB + col] = acc[mf][nf][r] * cs;
    }
  }
}

extern "C" void kernel_launch(void* const* d_in, const int* in_sizes, int n_in,
                              void* d_out, int out_size, void* d_ws,
                              size_t ws_size, hipStream_t stream) {
  (void)in_sizes; (void)n_in; (void)out_size; (void)ws_size;
  const float* hs = (const float*)d_in[0];
  const float* W  = (const float*)d_in[1];
  float* out = (float*)d_out;

  char* p = (char*)d_ws;
  unsigned char* Wq = (unsigned char*)p; p += (size_t)VOCAB * HID;
  unsigned char* Xq = (unsigned char*)p; p += (size_t)TOK * HID;
  float* sw = (float*)p;  p += (size_t)VOCAB * sizeof(float);
  float* cm = (float*)p;  p += (size_t)NCHUNK * HID * sizeof(float);
  float* tm = (float*)p;  p += (size_t)NCHUNK * sizeof(float);

  wquant_kernel<<<VOCAB, 256, 0, stream>>>(W, Wq, sw);
  cmax_kernel<<<dim3(NCHUNK, HID / 1024), 256, 0, stream>>>(hs, cm);
  tmax_kernel<<<NCHUNK, 256, 0, stream>>>(cm, tm);
  xquant_kernel<<<dim3(NCHUNK, HID / 1024), 256, 0, stream>>>(hs, cm, tm, Xq);
  gemm_bf8_kernel<<<dim3(TOK / BM, VOCAB / BN), 512, 0, stream>>>(Xq, Wq, sw,
                                                                  tm, out);
}
